// StableWaveEngine_32822140076294
// MI455X (gfx1250) — compile-verified, matrix operand order checked by reading, not run on hardware
//
#include <hip/hip_runtime.h>
#include <math.h>

typedef _Float16 v16h __attribute__((ext_vector_type(16)));
typedef _Float16 v8h  __attribute__((ext_vector_type(8)));
typedef float    v8f  __attribute__((ext_vector_type(8)));
typedef unsigned int u32x4 __attribute__((ext_vector_type(4)));
typedef int          i32x4 __attribute__((ext_vector_type(4)));
typedef int          i32x8 __attribute__((ext_vector_type(8)));

#define WAVE 32
constexpr int B = 2, S = 2048, D = 1024, F = 64, T = 28, V = 32000;
constexpr int M = B * S;  // 4096 rows
constexpr float RMS_EPS = 1.1920929e-07f;
constexpr float MAX_MAG = 50.0f;

// GEMM tiling: block tile 128x128, 8 waves (4x2), wave tile 32x64,
// K staged in KB=32 chunks, double-buffered LDS filled by the TDM.
constexpr int BM = 128, BN = 128, KB = 32;

// ---------------- embedding gather ----------------
__global__ void k_embed(const int* __restrict__ tok, const float* __restrict__ tab,
                        float* __restrict__ x) {
  int row = blockIdx.x;
  int t = tok[row];
  for (int d = threadIdx.x; d < D; d += blockDim.x)
    x[(size_t)row * D + d] = tab[(size_t)t * D + d];
}

// ---------------- f32 -> f16 conversion ----------------
__global__ void k_cvt16(const float* __restrict__ a, _Float16* __restrict__ o, int n) {
  int i = blockIdx.x * blockDim.x + threadIdx.x;
  if (i < n) o[i] = (_Float16)a[i];
}

// ---------------- RMSNorm (f32 in, f16 out) ----------------
__global__ void k_rmsnorm16(const float* __restrict__ x, const float* __restrict__ w,
                            _Float16* __restrict__ h) {
  __shared__ float red[256];
  int row = blockIdx.x;
  float s = 0.f;
  for (int d = threadIdx.x; d < D; d += 256) {
    float v = x[(size_t)row * D + d];
    s += v * v;
  }
  red[threadIdx.x] = s;
  __syncthreads();
  for (int off = 128; off; off >>= 1) {
    if (threadIdx.x < off) red[threadIdx.x] += red[threadIdx.x + off];
    __syncthreads();
  }
  float inv = rsqrtf(red[0] / (float)D + RMS_EPS);
  for (int d = threadIdx.x; d < D; d += 256)
    h[(size_t)row * D + d] = (_Float16)(x[(size_t)row * D + d] * inv * w[d]);
}

// ---------------- Tensor Data Mover: 2-D tile (rows x KB halves) -> LDS ----------------
// Builds the D# per CDNA5 ISA (CDNA5 ISA) and issues
// tensor_load_to_lds. Global tile start = gbase; row stride = K halves.
// LDS receives a compact [tile_rows][KB] row-major f16 tile at lds_off.
__device__ inline void tdm_load_2d(const _Float16* gbase, unsigned lds_off,
                                   int tensor_rows, int K, int tile_rows) {
  unsigned long long ga = (unsigned long long)(uintptr_t)gbase;
  u32x4 g0;
  g0[0] = 1u;                                           // count=1, user mode
  g0[1] = lds_off;                                      // LDS byte address
  g0[2] = (unsigned)(ga & 0xFFFFFFFFu);                 // global_addr[31:0]
  g0[3] = ((unsigned)(ga >> 32) & 0x1FFFFFFu) | (2u << 30);  // addr[56:32] | type=2

  unsigned td0 = (unsigned)K;           // tensor_dim0 (elements)
  unsigned td1 = (unsigned)tensor_rows; // tensor_dim1 (rows)
  unsigned st0 = (unsigned)K;           // tensor_dim0_stride (elements)
  i32x8 g1;
  g1[0] = (int)(1u << 16);                                   // data_size = 2 bytes
  g1[1] = (int)((td0 & 0xFFFFu) << 16);                      // tensor_dim0[15:0] @ bits63:48
  g1[2] = (int)((td0 >> 16) | ((td1 & 0xFFFFu) << 16));      // dim0[31:16], dim1[15:0]
  g1[3] = (int)((td1 >> 16) | ((unsigned)KB << 16));         // dim1[31:16], tile_dim0=KB
  g1[4] = (int)((unsigned)tile_rows);                        // tile_dim1; tile_dim2=0
  g1[5] = (int)st0;                                          // tensor_dim0_stride[31:0]
  g1[6] = 0;                                                 // stride[47:32], dim1_stride lo
  g1[7] = 0;
  i32x4 z = {};
#if __clang_major__ >= 23
  i32x8 z8 = {};
  __builtin_amdgcn_tensor_load_to_lds(g0, g1, z, z, z8, 0);
#else
  __builtin_amdgcn_tensor_load_to_lds(g0, g1, z, z, 0);
#endif
}

// ---------------- LDS-tiled WMMA GEMM: C[M,N] = A[M,K] * W[N,K]^T ----------------
// Double-buffered TDM pipeline: wave 0 issues the next stage's two descriptors
// before waiting on the current stage (TENSORcnt completes in order, so
// s_wait_tensorcnt 2 releases the current tile while the next is in flight).
// Optional epilogue: Cout = Acc + scale_p[scale_idx] * (A*W^T).
__global__ void k_gemm_tiled(const _Float16* __restrict__ A, const _Float16* __restrict__ Bw,
                             float* __restrict__ Cout, const float* __restrict__ Acc,
                             const float* __restrict__ scale_p, int scale_idx,
                             int Nn, int K) {
  __shared__ __align__(16) _Float16 ldsA[2][BM * KB];
  __shared__ __align__(16) _Float16 ldsB[2][BN * KB];
  const int row0 = blockIdx.x * BM;
  const int col0 = blockIdx.y * BN;
  const int wid = threadIdx.x / WAVE;        // 0..7
  const int lane = threadIdx.x & (WAVE - 1);
  const int r = lane & 15, hi = lane >> 4;
  const int wm = wid >> 1, wn = wid & 1;     // 4x2 wave grid -> wave tile 32x64

  const int nstages = K / KB;

  // prologue: stage 0 in flight
  if (threadIdx.x < WAVE) {
    tdm_load_2d(A + (size_t)row0 * K, (unsigned)(uintptr_t)&ldsA[0][0], BM, K, BM);
    tdm_load_2d(Bw + (size_t)col0 * K, (unsigned)(uintptr_t)&ldsB[0][0], BN, K, BN);
  }

  v8f acc[2][4] = {};

  for (int st = 0; st < nstages; ++st) {
    const int buf = st & 1;
    if (threadIdx.x < WAVE) {
      if (st + 1 < nstages) {
        int kb = (st + 1) * KB;
        tdm_load_2d(A + (size_t)row0 * K + kb,
                    (unsigned)(uintptr_t)&ldsA[buf ^ 1][0], BM, K, BM);
        tdm_load_2d(Bw + (size_t)col0 * K + kb,
                    (unsigned)(uintptr_t)&ldsB[buf ^ 1][0], BN, K, BN);
        __builtin_amdgcn_s_wait_tensorcnt(2);  // current tile done; next in flight
      } else {
        __builtin_amdgcn_s_wait_tensorcnt(0);
      }
    }
    __syncthreads();  // current tile visible to all waves

    v16h a[2], b[4];
#pragma unroll
    for (int mi = 0; mi < 2; mi++) {
      const _Float16* p = &ldsA[buf][(wm * 32 + mi * 16 + r) * KB + hi * 8];
      v8h lo = *(const v8h*)p;
      v8h up = *(const v8h*)(p + 16);
#pragma unroll
      for (int i = 0; i < 8; i++) { a[mi][i] = lo[i]; a[mi][i + 8] = up[i]; }
    }
#pragma unroll
    for (int ni = 0; ni < 4; ni++) {
      const _Float16* p = &ldsB[buf][(wn * 64 + ni * 16 + r) * KB + hi * 8];
      v8h lo = *(const v8h*)p;
      v8h up = *(const v8h*)(p + 16);
#pragma unroll
      for (int i = 0; i < 8; i++) { b[ni][i] = lo[i]; b[ni][i + 8] = up[i]; }
    }
#pragma unroll
    for (int mi = 0; mi < 2; mi++)
#pragma unroll
      for (int ni = 0; ni < 4; ni++)
        acc[mi][ni] = __builtin_amdgcn_wmma_f32_16x16x32_f16(
            false, a[mi], false, b[ni], (short)0, acc[mi][ni], false, false);

    __syncthreads();  // all reads of this buffer done before it is refilled
  }

  float sc = scale_p ? scale_p[scale_idx] : 1.0f;
#pragma unroll
  for (int mi = 0; mi < 2; mi++)
#pragma unroll
    for (int ni = 0; ni < 4; ni++)
#pragma unroll
      for (int v = 0; v < 8; v++) {
        size_t idx = (size_t)(row0 + wm * 32 + mi * 16 + v + hi * 8) * (size_t)Nn +
                     (col0 + wn * 64 + ni * 16 + r);
        float o = acc[mi][ni][v] * sc;
        if (Acc) o += Acc[idx];
        Cout[idx] = o;
      }
}

// ---------------- build complex waves with positional phase ----------------
__global__ void k_build_waves(const float* __restrict__ wc, const float* __restrict__ pos_freq,
                              float2* __restrict__ wbuf) {
  int bf = blockIdx.x;
  int b = bf / F, f = bf % F;
  float pf = pos_freq[f];
  for (int s = threadIdx.x; s < S; s += 256) {
    size_t row = (size_t)(b * S + s);
    float re = wc[row * (2 * F) + f];
    float im = wc[row * (2 * F) + F + f];
    float sn, c;
    sincosf((float)s * pf, &sn, &c);
    wbuf[(size_t)bf * S + s] = make_float2(re * c - im * sn, re * sn + im * c);
  }
}

// ---------------- radix-2 FFT core in LDS (length S=2048) ----------------
__device__ inline void fft_core(float2* ld, int t, float sign) {
  for (int len = 2; len <= S; len <<= 1) {
    int half = len >> 1;
    for (int j = t; j < S / 2; j += 256) {
      int grp = j / half, pos = j - grp * half;
      int i0 = grp * len + pos, i1 = i0 + half;
      float ang = sign * -6.28318530717958647692f * (float)pos / (float)len;
      float sn, c;
      sincosf(ang, &sn, &c);
      float2 u = ld[i0], w = ld[i1];
      float2 tw = make_float2(w.x * c - w.y * sn, w.x * sn + w.y * c);
      ld[i0] = make_float2(u.x + tw.x, u.y + tw.y);
      ld[i1] = make_float2(u.x - tw.x, u.y - tw.y);
    }
    __syncthreads();
  }
}

// forward FFT + fused per-frequency factor (phase*dampen*shift)
__global__ void k_fft_fwd(float2* __restrict__ wbuf, const float* __restrict__ speed,
                          const float* __restrict__ dampen, const float* __restrict__ phase_shift) {
  __shared__ float2 ld[S];
  int bf = blockIdx.x;
  int f = bf % F;
  for (int i = threadIdx.x; i < S; i += 256) {
    int rev = __brev((unsigned)i) >> (32 - 11);
    ld[rev] = wbuf[(size_t)bf * S + i];
  }
  __syncthreads();
  fft_core(ld, threadIdx.x, 1.0f);
  float xdp = dampen[f] + 0.5f;
  float softp = (xdp > 20.f) ? xdp : log1pf(expf(xdp));
  float damp = expf(-softp);
  float ang = 3.14159265358979323846f * speed[f] + phase_shift[f];
  float sn, c;
  sincosf(ang, &sn, &c);
  float fr = damp * c, fi = damp * sn;
  for (int i = threadIdx.x; i < S; i += 256) {
    float2 v = ld[i];
    wbuf[(size_t)bf * S + i] = make_float2(v.x * fr - v.y * fi, v.x * fi + v.y * fr);
  }
}

// ---------------- normalize coupling matrix ----------------
__global__ void k_cnorm(const float* __restrict__ cr, const float* __restrict__ ci,
                        float2* __restrict__ Cn) {
  __shared__ float red[256];
  float s = 0.f;
  for (int i = threadIdx.x; i < F * F; i += 256) s += cr[i] * cr[i] + ci[i] * ci[i];
  red[threadIdx.x] = s;
  __syncthreads();
  for (int off = 128; off; off >>= 1) {
    if (threadIdx.x < off) red[threadIdx.x] += red[threadIdx.x + off];
    __syncthreads();
  }
  float denom = fmaxf(sqrtf(red[0]) / (float)F, 1.0f);
  for (int i = threadIdx.x; i < F * F; i += 256)
    Cn[i] = make_float2(cr[i] / denom, ci[i] / denom);
}

// ---------------- coupling einsum + magnitude clamp ----------------
__global__ void k_couple(const float2* __restrict__ wf, const float2* __restrict__ Cn,
                         float2* __restrict__ out) {
  int idx = blockIdx.x * blockDim.x + threadIdx.x;  // B*F*S threads, s fastest
  int s = idx % S;
  int bg = idx / S;
  int g = bg % F, b = bg / F;
  float ar = 0.f, ai = 0.f;
  for (int f = 0; f < F; f++) {
    float2 w = wf[(size_t)(b * F + f) * S + s];
    float2 c = Cn[f * F + g];
    ar += w.x * c.x - w.y * c.y;
    ai += w.x * c.y + w.y * c.x;
  }
  float mag = sqrtf(ar * ar + ai * ai);
  float scl = fminf(MAX_MAG / (mag + 1e-8f), 1.0f);
  out[(size_t)(b * F + g) * S + s] = make_float2(ar * scl, ai * scl);
}

// inverse FFT, take real part, write f16 [M, F]
__global__ void k_fft_inv(const float2* __restrict__ wbuf2, _Float16* __restrict__ wr16) {
  __shared__ float2 ld[S];
  int bf = blockIdx.x;
  int b = bf / F, g = bf % F;
  for (int i = threadIdx.x; i < S; i += 256) {
    int rev = __brev((unsigned)i) >> (32 - 11);
    ld[rev] = wbuf2[(size_t)bf * S + i];
  }
  __syncthreads();
  fft_core(ld, threadIdx.x, -1.0f);
  const float invn = 1.0f / (float)S;
  for (int s = threadIdx.x; s < S; s += 256)
    wr16[(size_t)(b * S + s) * F + g] = (_Float16)(ld[s].x * invn);
}

// ---------------- host orchestration ----------------
extern "C" void kernel_launch(void* const* d_in, const int* in_sizes, int n_in,
                              void* d_out, int out_size, void* d_ws, size_t ws_size,
                              hipStream_t stream) {
  const int*   tokens      = (const int*)d_in[0];
  const float* embed       = (const float*)d_in[1];
  const float* norm_w      = (const float*)d_in[2];   // [T,D]
  const float* to_w        = (const float*)d_in[3];   // [2F,D]
  const float* from_w      = (const float*)d_in[4];   // [D,F]
  const float* step_scale  = (const float*)d_in[5];   // [T]
  const float* pos_freq    = (const float*)d_in[6];   // [F]
  const float* speed       = (const float*)d_in[7];
  const float* dampen      = (const float*)d_in[8];
  const float* phase_shift = (const float*)d_in[9];
  const float* cr          = (const float*)d_in[10];
  const float* ci          = (const float*)d_in[11];
  const float* out_norm    = (const float*)d_in[12];  // [D]
  const float* lm_w        = (const float*)d_in[13];  // [V,D]
  float* out = (float*)d_out;

  char* ws = (char*)d_ws;
  size_t off = 0;
  auto alloc = [&](size_t bytes) -> char* {
    char* p = ws + off;
    off = (off + bytes + 255) & ~(size_t)255;
    return p;
  };
  float*     x      = (float*)alloc((size_t)M * D * 4);
  _Float16*  h16    = (_Float16*)alloc((size_t)M * D * 2);
  float*     wc     = (float*)alloc((size_t)M * 2 * F * 4);
  float2*    wbuf   = (float2*)alloc((size_t)B * F * S * 8);
  float2*    wbuf2  = (float2*)alloc((size_t)B * F * S * 8);
  _Float16*  wr16   = (_Float16*)alloc((size_t)M * F * 2);
  _Float16*  to16   = (_Float16*)alloc((size_t)2 * F * D * 2);
  _Float16*  from16 = (_Float16*)alloc((size_t)D * F * 2);
  float2*    Cn     = (float2*)alloc((size_t)F * F * 8);
  _Float16*  lm16   = (_Float16*)alloc((size_t)V * D * 2);

  // one-time (per launch) prep
  k_embed<<<M, 256, 0, stream>>>(tokens, embed, x);
  k_cvt16<<<(2 * F * D + 255) / 256, 256, 0, stream>>>(to_w, to16, 2 * F * D);
  k_cvt16<<<(D * F + 255) / 256, 256, 0, stream>>>(from_w, from16, D * F);
  k_cvt16<<<(V * D + 255) / 256, 256, 0, stream>>>(lm_w, lm16, V * D);
  k_cnorm<<<1, 256, 0, stream>>>(cr, ci, Cn);

  for (int t = 0; t < T; t++) {
    k_rmsnorm16<<<M, 256, 0, stream>>>(x, norm_w + (size_t)t * D, h16);

    // wc[M,2F] = h16[M,D] @ to_w[2F,D]^T
    {
      dim3 grid(M / BM, (2 * F) / BN);
      k_gemm_tiled<<<grid, 256, 0, stream>>>(h16, to16, wc, nullptr, nullptr, 0,
                                             2 * F, D);
    }
    k_build_waves<<<B * F, 256, 0, stream>>>(wc, pos_freq, wbuf);
    k_fft_fwd<<<B * F, 256, 0, stream>>>(wbuf, speed, dampen, phase_shift);
    k_couple<<<(B * F * S) / 256, 256, 0, stream>>>(wbuf, Cn, wbuf2);
    k_fft_inv<<<B * F, 256, 0, stream>>>(wbuf2, wr16);

    // x = x + (wr16[M,F] @ from_w[D,F]^T) * step_scale[t]
    {
      dim3 grid(M / BM, D / BN);
      k_gemm_tiled<<<grid, 256, 0, stream>>>(wr16, from16, x, x, step_scale, t,
                                             D, F);
    }
  }

  k_rmsnorm16<<<M, 256, 0, stream>>>(x, out_norm, h16);
  // out[M,V] = h16[M,D] @ lm_w[V,D]^T
  {
    dim3 grid(M / BM, V / BN);
    k_gemm_tiled<<<grid, 256, 0, stream>>>(h16, lm16, out, nullptr, nullptr, 0,
                                           V, D);
  }
}
